// gated_tpp_67173288509868
// MI455X (gfx1250) — matrix operand order repaired and checked by reading, not compiled, not run on hardware
//
#include <hip/hip_runtime.h>
#include <hip/hip_bf16.h>
#include <math.h>

typedef __attribute__((ext_vector_type(16))) _Float16 v16h;
typedef __attribute__((ext_vector_type(8)))  float    v8f;
typedef __attribute__((ext_vector_type(4)))  unsigned int uint32x4;
typedef __attribute__((ext_vector_type(8)))  int      int32x8;
typedef __attribute__((ext_vector_type(4)))  int      int32x4;

#define BB   32
#define LL   512
#define DMOD 128
#define DTYP 4
#define NTYP 20
#define NSMP 50
#define HH   132
#define HN   144    // H padded to 9 N-tiles of 16
#define KN   160    // noise K padded to 5 k-chunks of 32
#define LOG10K 9.2103403719761840f

#if __has_builtin(__builtin_amdgcn_tensor_load_to_lds)
#define HAVE_TDM 1
#endif

__device__ __forceinline__ float frcp(float x)  { return __builtin_amdgcn_rcpf(x); }
__device__ __forceinline__ float frsq(float x)  { return __builtin_amdgcn_rsqf(x); }

__device__ __forceinline__ unsigned hash32(unsigned x) {
    x ^= x >> 17; x *= 0xed5ad4bbu;
    x ^= x >> 11; x *= 0xac4c1b51u;
    x ^= x >> 15; x *= 0x31848babu;
    x ^= x >> 14;
    return x;
}

__device__ __forceinline__ float softplusf(float y) {
    return (y > 15.f) ? y : __logf(1.f + __expf(y));
}

// ---------------------------------------------------------------------------
// Kernel A: per-event precompute.  hv = [sin|cos|type_emb] stored TRANSPOSED
// as f16 hvT[b][n][l] (n-major) so attention B-fragments are contiguous
// 32-byte spans in K=j.  Also the 4 pairwise-kernel scalar projections.
// ---------------------------------------------------------------------------
__global__ void precompute_kernel(const int* __restrict__ etype,
                                  const float* __restrict__ etime,
                                  const float* __restrict__ Wt,
                                  const float* __restrict__ temb,
                                  const float* __restrict__ Wg,
                                  const float* __restrict__ Wl,
                                  float* __restrict__ dgj, float* __restrict__ dgi,
                                  float* __restrict__ dlj, float* __restrict__ dli,
                                  _Float16* __restrict__ hvT)
{
    int idx = blockIdx.x * blockDim.x + threadIdx.x;
    if (idx >= BB * LL) return;
    int b = idx / LL, l = idx % LL;

    float t  = etime[idx];
    int   et = etype[idx];
    float te[DTYP];
    float gj = 0.f, gi2 = 0.f, lj = 0.f, li2 = 0.f;
    #pragma unroll
    for (int d = 0; d < DTYP; d++) {
        te[d] = temb[et * DTYP + d];
        gj  += te[d] * Wg[d];
        gi2 += te[d] * Wg[DTYP + d];
        lj  += te[d] * Wl[d];
        li2 += te[d] * Wl[DTYP + d];
    }
    dgj[idx] = gj; dgi[idx] = gi2; dlj[idx] = lj; dli[idx] = li2;

    float hv[HH];
    #pragma unroll 4
    for (int k = 0; k < DMOD / 2; k++) {
        float dv  = __expf(-(float)(2 * k) * (LOG10K / 128.f));
        float ang = (float)l * dv + t * Wt[k];
        float s, c;
        __sincosf(ang, &s, &c);
        hv[k] = s; hv[64 + k] = c;
    }
    #pragma unroll
    for (int d = 0; d < DTYP; d++) hv[DMOD + d] = te[d];

    _Float16* base = hvT + (size_t)b * HN * LL + l;
    for (int n = 0; n < HN; n++)
        base[(size_t)n * LL] = (_Float16)(n < HH ? hv[n] : 0.f);
}

// ---------------------------------------------------------------------------
// Kernel W: transposed padded f16 decoder noise weight WnT[n][k] (144x160).
// ---------------------------------------------------------------------------
__global__ void weightt_kernel(const float* __restrict__ Wnoise,
                               _Float16* __restrict__ WnT)
{
    int idx = blockIdx.x * blockDim.x + threadIdx.x;
    if (idx >= HN * KN) return;
    int n = idx / KN, k = idx % KN;
    float v = (n < HH && k < HH) ? Wnoise[n * HH + k] : 0.f;
    WnT[idx] = (_Float16)v;
}

// ---------------------------------------------------------------------------
// Kernel B: attention aggregation + LayerNorm, one wave per (b, 16-row tile).
// Scores built on the fly into the f16 A fragment; B fragments are aligned
// 32-byte vector loads from hvT; 9 f32 accumulator tiles cover H=132.
// ---------------------------------------------------------------------------
__global__ __launch_bounds__(32) void attn_ln_kernel(
        const float* __restrict__ etime,
        const float* __restrict__ dgj, const float* __restrict__ dgi,
        const float* __restrict__ dlj, const float* __restrict__ dli,
        const _Float16* __restrict__ hvT,
        const float* __restrict__ gamma, const float* __restrict__ beta,
        const float* __restrict__ bgp, const float* __restrict__ blp,
        float* __restrict__ h32)
{
    __shared__ float sh_t[LL], sh_g[LL], sh_l[LL];
    __shared__ float sh_hid[16][HN];

    const int lane = threadIdx.x;
    const int i0 = blockIdx.x * 16;
    const int b  = blockIdx.y;
    const float bg = bgp[0], bl = blp[0];

    for (int j = lane; j < LL; j += 32) {
        sh_t[j] = etime[b * LL + j];
        sh_g[j] = dgj[b * LL + j];
        sh_l[j] = dlj[b * LL + j];
    }
    __syncthreads();

    const int row = lane & 15;
    const int i   = i0 + row;
    const float ti = sh_t[i];
    const float gi = dgi[b * LL + i];
    const float li = dli[b * LL + i];
    const int   ksel = (lane < 16) ? 0 : 16;

    v8f acc[9] = {};
    const _Float16* hvb = hvT + (size_t)b * HN * LL;

    for (int j0 = 0; j0 < i0 + 16; j0 += 32) {
        // --- A fragment: 16(i) x 32(j) causal-masked scores ---
        v16h a;
        #pragma unroll
        for (int m = 0; m < 16; m++) {
            int kl = m + ksel;
            int j  = j0 + kl;
            float sc = 0.f;
            if (j < i) {
                float g  = frcp(1.f + __expf(-(sh_g[j] + gi + bg)));
                float ls = softplusf(sh_l[j] + li + bl) + 1e-6f;
                float td = fabsf(sh_t[j] - ti);
                float q  = 2.f * ls * ls;
                // gate / (1 + td^2/(2 ls^2)) == gate * q / (q + td^2)
                sc = g * q * frcp(q + td * td);
            }
            a[m] = (_Float16)sc;
        }
        // --- 9 N-tiles of B: aligned 32B vector loads, always in-bounds ---
        const int krow0 = j0 + ((lane < 16) ? 0 : 16);
        #pragma unroll
        for (int n = 0; n < 9; n++) {
            const _Float16* p = hvb + (size_t)(n * 16 + (lane & 15)) * LL + krow0;
            v16h bfrag = *(const v16h*)p;
            acc[n] = __builtin_amdgcn_wmma_f32_16x16x32_f16(
                         false, a, false, bfrag, (short)0, acc[n], false, false);
        }
    }

    // --- spill C tiles to LDS (vgpr r -> row r / r+8 per half-wave) ---
    #pragma unroll
    for (int n = 0; n < 9; n++)
        #pragma unroll
        for (int r = 0; r < 8; r++)
            sh_hid[(lane < 16) ? r : r + 8][n * 16 + (lane & 15)] = acc[n][r];
    __syncthreads();

    // --- LayerNorm per row over H=132 ---
    for (int rr = 0; rr < 16; rr++) {
        float s = 0.f, s2 = 0.f;
        for (int c = lane; c < HH; c += 32) {
            float v = sh_hid[rr][c];
            s += v; s2 += v * v;
        }
        #pragma unroll
        for (int m = 16; m >= 1; m >>= 1) {
            s  += __shfl_xor(s,  m, 32);
            s2 += __shfl_xor(s2, m, 32);
        }
        float mean = s * (1.f / (float)HH);
        float var  = s2 * (1.f / (float)HH) - mean * mean;
        float inv  = frsq(var + 1e-6f);
        size_t erow = (size_t)b * LL + i0 + rr;
        for (int c = lane; c < HH; c += 32)
            h32[erow * HH + c] = (sh_hid[rr][c] - mean) * inv * gamma[c] + beta[c];
    }
}

// ---------------------------------------------------------------------------
// Kernel C: decoder.  256 threads = 8 waves = 8 events per block.
// Wave 0 DMAs WnT (46KB) into LDS via the Tensor Data Mover, overlapped with
// per-event VALU work (u = hidden @ W_in^T, mark-prob logits).  Then each
// wave runs the sample GEMM with in-register PRNG noise A fragments against
// LDS B fragments; u is added per column before the ReLU.
// ---------------------------------------------------------------------------
__global__ __launch_bounds__(256) void decode_kernel(
        const float* __restrict__ h32, const _Float16* __restrict__ WnT,
        const float* __restrict__ Win,
        const float* __restrict__ w_time, const float* __restrict__ Wpred,
        float* __restrict__ out_mean, float* __restrict__ out_probs)
{
    __shared__ __attribute__((aligned(32))) _Float16 shW[HN * KN];  // 46080 B
    __shared__ float sh_h[8][HH];
    __shared__ float sh_u[8][HH];
    __shared__ float sh_logit[8][NTYP];

    const int tid  = threadIdx.x;
    const int lane = tid & 31;
    const int w    = tid >> 5;            // wave id = local event id
    const int e0   = blockIdx.x * 8;
    const int e    = e0 + w;

    // stage the 8 hidden vectors
    for (int idx = tid; idx < 8 * HH; idx += 256) {
        int we = idx / HH, c = idx % HH;
        sh_h[we][c] = h32[(size_t)(e0 + we) * HH + c];
    }
    __syncthreads();

    // ---- wave 0: async DMA of WnT into LDS via TDM ----
    if (w == 0) {
#ifdef HAVE_TDM
        unsigned long long ga = (unsigned long long)(uintptr_t)WnT;
        unsigned lds_off = (unsigned)(uintptr_t)&shW[0];
        uint32x4 g0;
        g0[0] = 1u;                                   // count=1, user descriptor
        g0[1] = lds_off;                              // lds_addr
        g0[2] = (unsigned)(ga & 0xFFFFFFFFu);         // global_addr[31:0]
        g0[3] = (unsigned)((ga >> 32) & 0x1FFFFFFu)   // global_addr[56:32]
              | (2u << 30);                           // type=2 (image)
        int32x8 g1;
        g1[0] = 1 << 16;                              // data_size=1 (2 bytes)
        g1[1] = (int)(KN << 16);                      // tensor_dim0[15:0]
        g1[2] = (int)(HN << 16);                      // tensor_dim1[15:0]
        g1[3] = (int)(KN << 16);                      // tile_dim0
        g1[4] = (int)HN;                              // tile_dim1 (tile_dim2=0)
        g1[5] = (int)KN;                              // tensor_dim0_stride[31:0]
        g1[6] = 0;
        g1[7] = 0;
        int32x4 g2 = {0, 0, 0, 0};
        int32x4 g3 = {0, 0, 0, 0};
        int32x8 g4 = {0, 0, 0, 0, 0, 0, 0, 0};
        __builtin_amdgcn_tensor_load_to_lds(g0, g1, g2, g3, g4, 0);
#else
        // fallback: cooperative copy by wave 0
        const uint4* src = (const uint4*)WnT;
        uint4* dst = (uint4*)shW;
        for (int idx = lane; idx < HN * KN * 2 / 16; idx += 32) dst[idx] = src[idx];
#endif
    }

    // ---- overlapped VALU: u = hidden @ W_in^T, and mark-prob logits ----
    for (int c = lane; c < HH; c += 32) {
        float acc = 0.f;
        const float* wr = Win + (size_t)c * HH;
        for (int k = 0; k < HH; k++) acc += wr[k] * sh_h[w][k];
        sh_u[w][c] = acc;
    }
    if (lane < NTYP) {
        float acc = 0.f;
        const float* wr = Wpred + (size_t)lane * HH;
        for (int k = 0; k < HH; k++) acc += wr[k] * sh_h[w][k];
        sh_logit[w][lane] = acc;
    }

#ifdef HAVE_TDM
    if (w == 0) __builtin_amdgcn_s_wait_tensorcnt(0);
#endif
    __syncthreads();   // shW, sh_u, sh_logit all visible

    // ---- softmax over 20 types ----
    float mx = -1e30f;
    for (int c = 0; c < NTYP; c++) mx = fmaxf(mx, sh_logit[w][c]);
    float den = 0.f;
    for (int c = 0; c < NTYP; c++) den += __expf(sh_logit[w][c] - mx);
    float rden = frcp(den);
    if (lane < NTYP)
        out_probs[(size_t)e * NTYP + lane] = __expf(sh_logit[w][lane] - mx) * rden;

    // ---- sample GEMM: noise (PRNG, in-register) x WnT (LDS) ----
    const int colN = lane & 15;
    float wt[9], uc[9];
    #pragma unroll
    for (int n = 0; n < 9; n++) {
        int c = n * 16 + colN;
        wt[n] = (c < HH) ? w_time[c] : 0.f;
        uc[n] = (c < HH) ? sh_u[w][c] : 0.f;
    }
    const int kselA = (lane < 16) ? 0 : 16;
    const int rowS  = lane & 15;
    float msum = 0.f;

    for (int st = 0; st < 4; st++) {
        const int s = st * 16 + rowS;        // A-fragment row = sample index
        v8f acc[9] = {};
        for (int k0 = 0; k0 < KN; k0 += 32) {
            v16h a;
            #pragma unroll
            for (int m = 0; m < 16; m += 2) {   // 2 uniforms per hash
                int kl = m + kselA;
                int k  = k0 + kl;
                float va = 0.f, vb = 0.f;
                if (k < HH) {
                    unsigned hsh = hash32((((unsigned)e << 6) | (unsigned)s) * 96u
                                          + (unsigned)(k >> 1));
                    va = (float)(hsh & 0xFFFFu) * (1.f / 65536.f);
                    vb = (float)(hsh >> 16)     * (1.f / 65536.f);
                }
                a[m]     = (_Float16)va;
                a[m + 1] = (_Float16)vb;
            }
            const int krow0 = k0 + ((lane < 16) ? 0 : 16);
            #pragma unroll
            for (int n = 0; n < 9; n++) {
                const _Float16* p = shW + (n * 16 + colN) * KN + krow0;
                v16h bfrag = *(const v16h*)p;      // ds_load_b128 x2
                acc[n] = __builtin_amdgcn_wmma_f32_16x16x32_f16(
                             false, a, false, bfrag, (short)0, acc[n], false, false);
            }
        }
        // relu(acc + u) * w_time, reduce over H per 16-lane half, softplus
        #pragma unroll
        for (int r = 0; r < 8; r++) {
            float p = 0.f;
            #pragma unroll
            for (int n = 0; n < 9; n++) p += fmaxf(acc[n][r] + uc[n], 0.f) * wt[n];
            #pragma unroll
            for (int m = 8; m >= 1; m >>= 1) p += __shfl_xor(p, m, 16);
            int srow = st * 16 + ((lane < 16) ? r : r + 8);
            if ((lane & 15) == 0 && srow < NSMP) msum += softplusf(p);
        }
    }
    float other = __shfl_xor(msum, 16, 32);
    if (lane == 0) out_mean[e] = (msum + other) * (1.f / (float)NSMP);
}

// ---------------------------------------------------------------------------
extern "C" void kernel_launch(void* const* d_in, const int* in_sizes, int n_in,
                              void* d_out, int out_size, void* d_ws, size_t ws_size,
                              hipStream_t stream)
{
    const int*   etype  = (const int*)  d_in[0];
    const float* etime  = (const float*)d_in[1];
    // d_in[2] arrival_times unused by forward
    const float* Wt     = (const float*)d_in[3];
    const float* temb   = (const float*)d_in[4];
    const float* Wg     = (const float*)d_in[5];
    const float* bgp    = (const float*)d_in[6];
    const float* Wl     = (const float*)d_in[7];
    const float* blp    = (const float*)d_in[8];
    const float* gamma  = (const float*)d_in[9];
    const float* beta   = (const float*)d_in[10];
    const float* Win    = (const float*)d_in[11];
    const float* Wnoise = (const float*)d_in[12];
    const float* wtime  = (const float*)d_in[13];
    const float* Wpred  = (const float*)d_in[14];

    float* out_mean  = (float*)d_out;                   // [B,L]
    float* out_probs = (float*)d_out + (size_t)BB * LL; // [B,L,NTYP]

    char* ws = (char*)d_ws;
    size_t o = 0;
    float*    dgj = (float*)(ws + o); o += (size_t)BB * LL * 4;
    float*    dgi = (float*)(ws + o); o += (size_t)BB * LL * 4;
    float*    dlj = (float*)(ws + o); o += (size_t)BB * LL * 4;
    float*    dli = (float*)(ws + o); o += (size_t)BB * LL * 4;
    _Float16* hvT = (_Float16*)(ws + o); o += (size_t)BB * HN * LL * 2;
    float*    h32 = (float*)(ws + o);    o += (size_t)BB * LL * HH * 4;
    _Float16* WnT = (_Float16*)(ws + o); o += (size_t)HN * KN * 2;

    precompute_kernel<<<(BB * LL + 255) / 256, 256, 0, stream>>>(
        etype, etime, Wt, temb, Wg, Wl, dgj, dgi, dlj, dli, hvT);

    weightt_kernel<<<(HN * KN + 255) / 256, 256, 0, stream>>>(Wnoise, WnT);

    attn_ln_kernel<<<dim3(LL / 16, BB), 32, 0, stream>>>(
        etime, dgj, dgi, dlj, dli, hvT, gamma, beta, bgp, blp, h32);

    decode_kernel<<<(BB * LL) / 8, 256, 0, stream>>>(
        h32, WnT, Win, wtime, Wpred, out_mean, out_probs);
}
